// RPN_Detector_Ball_30992484008031
// MI455X (gfx1250) — matrix operand order repaired and checked by reading, not compiled, not run on hardware
//
#include <hip/hip_runtime.h>
#include <hip/hip_bf16.h>

// ---------------------------------------------------------------------------
// RPN detector (ball-query + conv stack + KNN fusion + MLP head) for gfx1250.
// All 1x1 convs are row-GEMMs executed with v_wmma_f32_16x16x32_f16
// (f16 A/B, fp32 accumulate). Each wave32 owns a 32x64 output macro-tile
// (2x4 register blocking = 8 WMMAs per K-step, 24 B/lane moved per WMMA).
// ---------------------------------------------------------------------------

#define B_ 4
#define N_ 16384
#define M_ 512

typedef __attribute__((ext_vector_type(8)))  _Float16 v8h;
typedef __attribute__((ext_vector_type(16))) _Float16 v16h;
typedef __attribute__((ext_vector_type(8)))  float    v8f;

static __device__ inline v16h load_frag(const _Float16* p, int gap)
{
    v8h lo = *(const v8h*)(p);
    v8h hi = *(const v8h*)(p + gap);
    return __builtin_shufflevector(lo, hi, 0,1,2,3,4,5,6,7,8,9,10,11,12,13,14,15);
}

// ---------------------------------------------------------------------------
// f16 GEMM + bias + relu.  A: [R x K] row-major f16 (K mult of 32, R mult 32),
// W: [Cout x K] row-major f16 (Cout mult 64), Y: [R x Cout] f16.
// Lane layouts used here (A and B both in contiguous k order):
//   A 16x32 f16: lanes 0-15 rows 0-15; halves 0..15 -> K = kb16+0..15,
//                kb16 = (lane>>4)*16.
//   B 32x16 f16: lane = column; halves 0..15 -> K = kb16+0..15, kb16=(lane>>4)*16.
//   D 16x16 f32: lane = col (lanes 0-15 / 16-31), elem v -> row v + (lane>>4)*8.
// ---------------------------------------------------------------------------
__global__ __launch_bounds__(256) void k_gemm_relu(
    const _Float16* __restrict__ A, const _Float16* __restrict__ W,
    const float* __restrict__ bias, _Float16* __restrict__ Y,
    int R, int K, int Cout)
{
    const int lane = threadIdx.x & 31;
    const int wid  = (blockIdx.x * blockDim.x + threadIdx.x) >> 5;
    const int tilesN = Cout >> 6;   // 64-column macro tiles
    const int tilesM = R >> 5;      // 32-row macro tiles
    const int ti = wid / tilesN;
    const int tj = wid - ti * tilesN;
    if (ti >= tilesM) return;

    const int half = lane >> 4;     // 0 or 1
    const int l15  = lane & 15;

    const _Float16* ap0 = A + (size_t)(ti * 32 +      l15) * K + half * 16;
    const _Float16* ap1 = A + (size_t)(ti * 32 + 16 + l15) * K + half * 16;
    const _Float16* bp0 = W + (size_t)(tj * 64 +  0 + l15) * K + half * 16;
    const _Float16* bp1 = W + (size_t)(tj * 64 + 16 + l15) * K + half * 16;
    const _Float16* bp2 = W + (size_t)(tj * 64 + 32 + l15) * K + half * 16;
    const _Float16* bp3 = W + (size_t)(tj * 64 + 48 + l15) * K + half * 16;

    v8f acc[2][4];
#pragma unroll
    for (int r = 0; r < 2; ++r)
#pragma unroll
        for (int c = 0; c < 4; ++c)
            acc[r][c] = (v8f){0.f, 0.f, 0.f, 0.f, 0.f, 0.f, 0.f, 0.f};

    for (int k0 = 0; k0 < K; k0 += 32) {
        v16h a0 = load_frag(ap0 + k0, 8);
        v16h a1 = load_frag(ap1 + k0, 8);
        v16h b0 = load_frag(bp0 + k0, 8);
        v16h b1 = load_frag(bp1 + k0, 8);
        v16h b2 = load_frag(bp2 + k0, 8);
        v16h b3 = load_frag(bp3 + k0, 8);
        acc[0][0] = __builtin_amdgcn_wmma_f32_16x16x32_f16(false, a0, false, b0, (short)0, acc[0][0], false, false);
        acc[0][1] = __builtin_amdgcn_wmma_f32_16x16x32_f16(false, a0, false, b1, (short)0, acc[0][1], false, false);
        acc[0][2] = __builtin_amdgcn_wmma_f32_16x16x32_f16(false, a0, false, b2, (short)0, acc[0][2], false, false);
        acc[0][3] = __builtin_amdgcn_wmma_f32_16x16x32_f16(false, a0, false, b3, (short)0, acc[0][3], false, false);
        acc[1][0] = __builtin_amdgcn_wmma_f32_16x16x32_f16(false, a1, false, b0, (short)0, acc[1][0], false, false);
        acc[1][1] = __builtin_amdgcn_wmma_f32_16x16x32_f16(false, a1, false, b1, (short)0, acc[1][1], false, false);
        acc[1][2] = __builtin_amdgcn_wmma_f32_16x16x32_f16(false, a1, false, b2, (short)0, acc[1][2], false, false);
        acc[1][3] = __builtin_amdgcn_wmma_f32_16x16x32_f16(false, a1, false, b3, (short)0, acc[1][3], false, false);
    }

#pragma unroll
    for (int c = 0; c < 4; ++c) {
        const int colD = tj * 64 + c * 16 + l15;
        const float bb = bias[colD];
#pragma unroll
        for (int r = 0; r < 2; ++r) {
            _Float16* yp = Y + (size_t)(ti * 32 + r * 16) * Cout + colD;
#pragma unroll
            for (int v = 0; v < 8; ++v) {
                float rr = acc[r][c][v] + bb;
                rr = rr > 0.f ? rr : 0.f;
                yp[(size_t)(v + half * 8) * Cout] = (_Float16)rr;
            }
        }
    }
}

// Convert fp32 (cout,cin) weights -> f16 (cout,cinp) with zero K-padding.
__global__ void k_pack_w(const float* __restrict__ src, _Float16* __restrict__ dst,
                         int cout, int cin, int cinp)
{
    int t = blockIdx.x * blockDim.x + threadIdx.x;
    if (t >= cout * cinp) return;
    int o = t / cinp, k = t - o * cinp;
    dst[t] = (k < cin) ? (_Float16)src[o * cin + k] : (_Float16)0.f;
}

// Ball query: one wave per (b,m); ordered append of first 64 in-ball indices,
// pad remainder with first hit (0 if no hits) — matches top_k-on-index ref.
__global__ __launch_bounds__(256) void k_ball(
    const float* __restrict__ x, const float* __restrict__ node,
    int* __restrict__ idx)
{
    const int lane = threadIdx.x & 31;
    const int nm = (blockIdx.x * blockDim.x + threadIdx.x) >> 5;
    if (nm >= B_ * M_) return;
    const int b = nm >> 9, m = nm & 511;
    const float* nb = node + (size_t)b * 3 * M_;
    const float nx = nb[m], ny = nb[M_ + m], nz = nb[2 * M_ + m];
    const float* xb = x + (size_t)b * 3 * N_;
    int* out = idx + (size_t)nm * 64;

    int cnt = 0, firstIdx = 0;
    bool haveFirst = false;
    for (int n0 = 0; n0 < N_; n0 += 32) {
        int n = n0 + lane;
        float dx = xb[n] - nx, dy = xb[N_ + n] - ny, dz = xb[2 * N_ + n] - nz;
        float d2 = dx * dx + dy * dy + dz * dz;
        bool pred = d2 < 4.0f;
        unsigned mask = (unsigned)__ballot(pred);
        if (!haveFirst && mask) { firstIdx = n0 + __ffs(mask) - 1; haveFirst = true; }
        if (pred) {
            int pos = cnt + __popc(mask & ((1u << lane) - 1u));
            if (pos < 64) out[pos] = n;
        }
        cnt += __popc(mask);
        if (cnt >= 64) break;
    }
    if (cnt < 64)
        for (int j = cnt + lane; j < 64; j += 32) out[j] = firstIdx;
}

// Gather ball neighborhood rows: X0[row, 0:3]=x-node, [3:6]=sn, [6:32]=0 (f16).
__global__ void k_gather_ball(const float* __restrict__ x, const float* __restrict__ sn,
                              const float* __restrict__ node, const int* __restrict__ idx,
                              _Float16* __restrict__ X0)
{
    int row = blockIdx.x * blockDim.x + threadIdx.x;
    if (row >= B_ * M_ * 64) return;
    int bm = row >> 6, b = bm >> 9, m = bm & 511;
    int n = idx[row];
    const float* xb = x + (size_t)b * 3 * N_;
    const float* sb = sn + (size_t)b * 3 * N_;
    const float* nb = node + (size_t)b * 3 * M_;
    _Float16* o = X0 + (size_t)row * 32;
    o[0] = (_Float16)(xb[n] - nb[m]);
    o[1] = (_Float16)(xb[N_ + n] - nb[M_ + m]);
    o[2] = (_Float16)(xb[2 * N_ + n] - nb[2 * M_ + m]);
    o[3] = (_Float16)sb[n];
    o[4] = (_Float16)sb[N_ + n];
    o[5] = (_Float16)sb[2 * N_ + n];
    for (int c = 6; c < 32; ++c) o[c] = (_Float16)0.f;
}

// Per-node max over 64 points (64ch) + concat [f, fmax] -> 128ch.
__global__ void k_pool_concat64(const _Float16* __restrict__ Xin, _Float16* __restrict__ Xout)
{
    int bm = blockIdx.x, t = threadIdx.x; // 64 threads = channels
    const _Float16* in = Xin + (size_t)bm * 64 * 64;
    float mx = 0.f; // post-relu values are >= 0
    for (int p = 0; p < 64; ++p) mx = fmaxf(mx, (float)in[p * 64 + t]);
    _Float16 hm = (_Float16)mx;
    _Float16* o = Xout + (size_t)bm * 64 * 128;
    for (int p = 0; p < 64; ++p) { o[p * 128 + t] = in[p * 64 + t]; o[p * 128 + 64 + t] = hm; }
}

// feat1[bm, c] = max over 64 points (128ch).
__global__ void k_pool_feat1(const _Float16* __restrict__ Xin, _Float16* __restrict__ feat1)
{
    int bm = blockIdx.x, t = threadIdx.x; // 128 threads
    const _Float16* in = Xin + (size_t)bm * 64 * 128;
    float mx = 0.f;
    for (int p = 0; p < 64; ++p) mx = fmaxf(mx, (float)in[p * 128 + t]);
    feat1[(size_t)bm * 128 + t] = (_Float16)mx;
}

// KNN: one wave per (b,m), 9 rounds of wave-min on (distbits<<32 | idx) keys
// (tie-break by index = lax.top_k order). Self (d=0) is included, as in ref.
__global__ __launch_bounds__(256) void k_knn(const float* __restrict__ node,
                                             int* __restrict__ knn_idx)
{
    const int lane = threadIdx.x & 31;
    const int nm = (blockIdx.x * blockDim.x + threadIdx.x) >> 5;
    if (nm >= B_ * M_) return;
    const int b = nm >> 9, m = nm & 511;
    const float* nb = node + (size_t)b * 3 * M_;
    const float qx = nb[m], qy = nb[M_ + m], qz = nb[2 * M_ + m];

    float d2v[16];
#pragma unroll
    for (int s = 0; s < 16; ++s) {
        int j = lane + 32 * s;
        float dx = nb[j] - qx, dy = nb[M_ + j] - qy, dz = nb[2 * M_ + j] - qz;
        d2v[s] = dx * dx + dy * dy + dz * dz;
    }
    unsigned chosen = 0;
    for (int t = 0; t < 9; ++t) {
        unsigned long long best = ~0ULL;
#pragma unroll
        for (int s = 0; s < 16; ++s) {
            if (chosen & (1u << s)) continue;
            unsigned long long key =
                ((unsigned long long)__float_as_uint(d2v[s]) << 32) |
                (unsigned)(lane + 32 * s);
            if (key < best) best = key;
        }
        for (int off = 16; off >= 1; off >>= 1) {
            unsigned long long o = __shfl_xor(best, off, 32);
            if (o < best) best = o;
        }
        int widx = (int)(best & 0xffffffffu);
        if ((widx & 31) == lane) chosen |= 1u << (widx >> 5);
        if (lane == 0) knn_idx[nm * 16 + t] = widx;
    }
}

// Build KNN rows (K padded 9->16): [0:3]=node[nn]-node[m], [3:131]=feat1[nn], pad 0.
__global__ void k_gather_knn(const float* __restrict__ node, const _Float16* __restrict__ feat1,
                             const int* __restrict__ knn_idx, _Float16* __restrict__ Xk)
{
    int row = blockIdx.x, t = threadIdx.x; // 160 threads = padded channels
    int j = row & 15, bm = row >> 4, b = bm >> 9, m = bm & 511;
    _Float16 v = (_Float16)0.f;
    if (j < 9) {
        int nn = knn_idx[bm * 16 + j];
        if (t < 3) {
            const float* nb = node + (size_t)b * 3 * M_;
            v = (_Float16)(nb[t * M_ + nn] - nb[t * M_ + m]);
        } else if (t < 131) {
            v = feat1[((size_t)b * M_ + nn) * 128 + (t - 3)];
        }
    }
    Xk[(size_t)row * 160 + t] = v;
}

// Max over j<9 (256ch) + concat [g, gmax] -> 512ch over all 16 padded rows.
__global__ void k_pool_concat_knn(const _Float16* __restrict__ Xin, _Float16* __restrict__ Xout)
{
    int bm = blockIdx.x, t = threadIdx.x; // 256 threads
    const _Float16* in = Xin + (size_t)bm * 16 * 256;
    float mx = 0.f;
    for (int j = 0; j < 9; ++j) mx = fmaxf(mx, (float)in[j * 256 + t]);
    _Float16 hm = (_Float16)mx;
    _Float16* o = Xout + (size_t)bm * 16 * 512;
    for (int j = 0; j < 16; ++j) { o[j * 512 + t] = in[j * 256 + t]; o[j * 512 + 256 + t] = hm; }
}

// feat2[bm, c] = max over j<9 (512ch).
__global__ void k_pool_feat2(const _Float16* __restrict__ Xin, _Float16* __restrict__ feat2)
{
    int bm = blockIdx.x, t = threadIdx.x; // 512 threads
    const _Float16* in = Xin + (size_t)bm * 16 * 512;
    float mx = 0.f;
    for (int j = 0; j < 9; ++j) mx = fmaxf(mx, (float)in[j * 512 + t]);
    feat2[(size_t)bm * 512 + t] = (_Float16)mx;
}

// H = concat([feat1, feat2]) -> 640ch per node.
__global__ void k_build_H(const _Float16* __restrict__ feat1, const _Float16* __restrict__ feat2,
                          _Float16* __restrict__ H)
{
    int bm = blockIdx.x, t = threadIdx.x; // 640 threads
    H[(size_t)bm * 640 + t] =
        (t < 128) ? feat1[(size_t)bm * 128 + t] : feat2[(size_t)bm * 512 + (t - 128)];
}

// mlp3 (256->4, fp32, no relu) + outputs: node copy, keypoints, softplus sigma.
__global__ void k_final(const _Float16* __restrict__ H2, const float* __restrict__ w3,
                        const float* __restrict__ b3, const float* __restrict__ node,
                        float* __restrict__ out)
{
    int bm = blockIdx.x * blockDim.x + threadIdx.x;
    if (bm >= B_ * M_) return;
    int b = bm >> 9, m = bm & 511;
    const _Float16* h = H2 + (size_t)bm * 256;
    float acc0 = b3[0], acc1 = b3[1], acc2 = b3[2], acc3 = b3[3];
    for (int k = 0; k < 256; ++k) {
        float hv = (float)h[k];
        acc0 += hv * w3[0 * 256 + k];
        acc1 += hv * w3[1 * 256 + k];
        acc2 += hv * w3[2 * 256 + k];
        acc3 += hv * w3[3 * 256 + k];
    }
    const float* nb = node + (size_t)b * 3 * M_;
    float n0 = nb[m], n1 = nb[M_ + m], n2 = nb[2 * M_ + m];
    float* o_node = out;
    float* o_kp   = out + B_ * 3 * M_;
    float* o_sg   = out + 2 * B_ * 3 * M_;
    o_node[b * 3 * M_ + 0 * M_ + m] = n0;
    o_node[b * 3 * M_ + 1 * M_ + m] = n1;
    o_node[b * 3 * M_ + 2 * M_ + m] = n2;
    o_kp[b * 3 * M_ + 0 * M_ + m] = acc0 + n0;
    o_kp[b * 3 * M_ + 1 * M_ + m] = acc1 + n1;
    o_kp[b * 3 * M_ + 2 * M_ + m] = acc2 + n2;
    float v = acc3; // stable softplus + lower bound
    o_sg[b * M_ + m] = fmaxf(v, 0.f) + log1pf(expf(-fabsf(v))) + 0.001f;
}

// ---------------------------------------------------------------------------
extern "C" void kernel_launch(void* const* d_in, const int* in_sizes, int n_in,
                              void* d_out, int out_size, void* d_ws, size_t ws_size,
                              hipStream_t stream)
{
    const float* x    = (const float*)d_in[0];
    const float* sn   = (const float*)d_in[1];
    const float* node = (const float*)d_in[2];
    const float* w[12];
    const float* bias[12];
    for (int i = 0; i < 12; ++i) {
        w[i]    = (const float*)d_in[3 + 2 * i];
        bias[i] = (const float*)d_in[4 + 2 * i];
    }
    const float* w3 = (const float*)d_in[27];
    const float* b3 = (const float*)d_in[28];

    // Workspace layout (256B aligned carve-outs).
    char* ws = (char*)d_ws;
    size_t off = 0;
    auto carve = [&](size_t bytes) -> void* {
        void* p = ws + off;
        off += (bytes + 255) & ~(size_t)255;
        return p;
    };
    _Float16* bufA   = (_Float16*)carve((size_t)131072 * 128 * 2); // 32 MB ping
    _Float16* bufB   = (_Float16*)carve((size_t)131072 * 128 * 2); // 32 MB pong
    int*      idxB   = (int*)     carve((size_t)131072 * 4);
    _Float16* feat1  = (_Float16*)carve((size_t)2048 * 128 * 2);
    int*      knnI   = (int*)     carve((size_t)2048 * 16 * 4);
    _Float16* feat2  = (_Float16*)carve((size_t)2048 * 512 * 2);
    _Float16* Hbuf   = (_Float16*)carve((size_t)2048 * 640 * 2);
    _Float16* H1     = (_Float16*)carve((size_t)2048 * 512 * 2);
    _Float16* H2     = (_Float16*)carve((size_t)2048 * 256 * 2);

    // Layer shapes: cout x cin (K-padded to cinp, multiple of 32).
    const int cout_[12] = {64, 64, 64, 128, 128, 256, 256, 256, 512, 512, 512, 256};
    const int cin_ [12] = { 6, 64, 64, 128, 128, 131, 256, 256, 512, 512, 640, 512};
    const int cinp_[12] = {32, 64, 64, 128, 128, 160, 256, 256, 512, 512, 640, 512};
    _Float16* Wp[12];
    for (int i = 0; i < 12; ++i) Wp[i] = (_Float16*)carve((size_t)cout_[i] * cinp_[i] * 2);

    for (int i = 0; i < 12; ++i) {
        int n = cout_[i] * cinp_[i];
        k_pack_w<<<(n + 255) / 256, 256, 0, stream>>>(w[i], Wp[i], cout_[i], cin_[i], cinp_[i]);
    }

    auto gemm = [&](const _Float16* A, int wi, _Float16* Y, int R) {
        int K = cinp_[wi], Cout = cout_[wi];
        int waves = (R / 32) * (Cout / 64);   // 32x64 macro-tiles
        k_gemm_relu<<<(waves + 7) / 8, 256, 0, stream>>>(A, Wp[wi], bias[wi], Y, R, K, Cout);
    };

    const int R1 = B_ * M_ * 64; // 131072 ball rows
    const int R2 = B_ * M_ * 16; // 32768 padded KNN rows
    const int R3 = B_ * M_;      // 2048 node rows

    // Stage 1: ball query + point conv stack -> feat1.
    k_ball<<<(B_ * M_) / 8, 256, 0, stream>>>(x, node, idxB);
    k_gather_ball<<<R1 / 256, 256, 0, stream>>>(x, sn, node, idxB, bufA);
    gemm(bufA, 0, bufB, R1);                               // conv1  6->64
    gemm(bufB, 1, bufA, R1);                               // conv2 64->64
    gemm(bufA, 2, bufB, R1);                               // conv3 64->64
    k_pool_concat64<<<R3, 64, 0, stream>>>(bufB, bufA);    // -> 128ch
    gemm(bufA, 3, bufB, R1);                               // conv4 128->128
    gemm(bufB, 4, bufA, R1);                               // conv5 128->128
    k_pool_feat1<<<R3, 128, 0, stream>>>(bufA, feat1);

    // Stage 2: node KNN fusion -> feat2.
    k_knn<<<(B_ * M_) / 8, 256, 0, stream>>>(node, knnI);
    k_gather_knn<<<R2, 160, 0, stream>>>(node, feat1, knnI, bufB);
    gemm(bufB, 5, bufA, R2);                               // knnb1 131->256
    gemm(bufA, 6, bufB, R2);                               // knnb2 256->256
    gemm(bufB, 7, bufA, R2);                               // knnb3 256->256
    k_pool_concat_knn<<<R3, 256, 0, stream>>>(bufA, bufB); // -> 512ch
    gemm(bufB, 8, bufA, R2);                               // knna1 512->512
    gemm(bufA, 9, bufB, R2);                               // knna2 512->512
    k_pool_feat2<<<R3, 512, 0, stream>>>(bufB, feat2);

    // Stage 3: head MLP + outputs.
    k_build_H<<<R3, 640, 0, stream>>>(feat1, feat2, Hbuf);
    gemm(Hbuf, 10, H1, R3);                                // mlp1 640->512
    gemm(H1, 11, H2, R3);                                  // mlp2 512->256
    k_final<<<R3 / 256, 256, 0, stream>>>(H2, w3, b3, node, (float*)d_out);
}
